// SimpleRNN_7859790151809
// MI455X (gfx1250) — matrix operand order repaired and checked by reading, not compiled, not run on hardware
//
#include <hip/hip_runtime.h>
#include <hip/hip_bf16.h>
#include <math.h>

// Problem sizes (fixed by the reference)
#define B_SZ 32
#define T_SZ 1024
#define D_SZ 128
#define H_SZ 512
#define C_SZ 10
#define M_ROWS (T_SZ * B_SZ)          // 32768 rows for the big GEMMs

typedef __attribute__((ext_vector_type(16))) __bf16 v16bf;
typedef __attribute__((ext_vector_type(8)))  __bf16 v8bf;
typedef __attribute__((ext_vector_type(8)))  float  v8f;
typedef __attribute__((ext_vector_type(4)))  float  v4f;

__device__ __forceinline__ v8f wmma_bf16(v16bf a, v16bf b, v8f c) {
  // D = A(16x32 bf16) * B(32x16 bf16) + C(16x16 f32)
  return __builtin_amdgcn_wmma_f32_16x16x32_bf16(
      /*neg_a=*/false, a, /*neg_b=*/false, b,
      /*c_mod=*/(short)0, c, /*reuse_a=*/false, /*reuse_b=*/false);
}

// Async global -> LDS copy, 16 bytes per lane (ASYNCcnt-tracked DMA path).
// VDST VGPR carries the LDS byte address, VADDR the 64-bit global address.
__device__ __forceinline__ void async_load_b128(void* lds_ptr, const void* gptr) {
  unsigned lds_off = (unsigned)(unsigned long long)lds_ptr;
  asm volatile("global_load_async_to_lds_b128 %0, %1, off"
               :: "v"(lds_off), "v"(gptr)
               : "memory");
}
__device__ __forceinline__ void wait_asynccnt0() {
  asm volatile("s_wait_asynccnt 0x0" ::: "memory");
}

// ---------------------------------------------------------------------------
// Pack a row-major f32 weight W[K][N] into bf16 "B-fragment" order:
//   Wp[strip s = n0/16][chunk c = k0/32][lane l 0..31][e 0..15]
// B 32x16 bf16 lane layout: kbase = (l<16) ? 0 : 8, k = kbase + e + (e>=8 ? 8 : 0)
// (matches A); n = s*16 + (l & 15).  One 32-byte load per fragment.
// ---------------------------------------------------------------------------
__global__ void pack_weight_bf16(const float* __restrict__ W,
                                 __bf16* __restrict__ Wp, int K, int N) {
  int i = blockIdx.x * blockDim.x + threadIdx.x;
  int total = K * N;
  if (i >= total) return;
  int chunks = K >> 5;                  // K/32
  int s   = i / (chunks * 512);
  int rem = i - s * (chunks * 512);
  int c   = rem >> 9;                   // /512
  int q   = rem & 511;
  int l   = q >> 4;
  int e   = q & 15;
  int k   = c * 32 + ((l < 16) ? 0 : 8) + e + ((e >= 8) ? 8 : 0);
  int n   = s * 16 + (l & 15);
  Wp[i] = (__bf16)W[(size_t)k * N + n];
}

// ---------------------------------------------------------------------------
// A-fragment build helpers. 16x32 bf16 A lane layout:
//   m = lane & 15 ; kbase = (lane<16) ? 0 : 8
//   element e -> k = kbase + e + (e>=8 ? 8 : 0)   (two contiguous 8-runs)
// ---------------------------------------------------------------------------
__device__ __forceinline__ v16bf a_frag_from_bf16(const __bf16* row, int k0, int kb) {
  v8bf lo = *(const v8bf*)(row + k0 + kb);
  v8bf hi = *(const v8bf*)(row + k0 + kb + 16);
  v16bf a;
#pragma unroll
  for (int e = 0; e < 8; ++e) { a[e] = lo[e]; a[e + 8] = hi[e]; }
  return a;
}

__device__ __forceinline__ v16bf a_frag_from_f32(const float* row, int k0, int kb) {
  v4f f0 = *(const v4f*)(row + k0 + kb);
  v4f f1 = *(const v4f*)(row + k0 + kb + 4);
  v4f f2 = *(const v4f*)(row + k0 + kb + 16);
  v4f f3 = *(const v4f*)(row + k0 + kb + 20);
  v16bf a;
#pragma unroll
  for (int e = 0; e < 4; ++e) {
    a[e]      = (__bf16)f0[e];
    a[e + 4]  = (__bf16)f1[e];
    a[e + 8]  = (__bf16)f2[e];
    a[e + 12] = (__bf16)f3[e];
  }
  return a;
}

// ---------------------------------------------------------------------------
// proj1: xw[t*32+b][n] = sum_d x[b][t][d] * Wx1[d][n] + b1[n]
// One 16x16 tile per wave, 8 waves/block, exact grid (EXEC all-ones for WMMA).
// ---------------------------------------------------------------------------
__global__ void __launch_bounds__(256)
proj1_kernel(const float* __restrict__ x, const __bf16* __restrict__ Wxp,
             const float* __restrict__ bias, float* __restrict__ xw) {
  int tid = threadIdx.x, lane = tid & 31, wave = tid >> 5;
  int tile = blockIdx.x * 8 + wave;     // (M_ROWS/16)*(H/16) = 65536 tiles
  int mt = tile >> 5;                   // / (H/16 = 32)
  int nt = tile & 31;
  int t  = mt >> 1;                     // 16 rows never straddle a t (B=32)
  int b0 = (mt & 1) * 16;
  int n0 = nt * 16;
  int nl = lane & 15;
  int kb = (lane < 16) ? 0 : 8;
  const float* arow = x + ((size_t)(b0 + nl) * T_SZ + t) * D_SZ;

  v8f acc = {};
#pragma unroll
  for (int c = 0; c < D_SZ / 32; ++c) {
    v16bf a = a_frag_from_f32(arow, c * 32, kb);
    v16bf b = *(const v16bf*)(Wxp + ((size_t)(nt * (D_SZ / 32) + c) * 512 + lane * 16));
    acc = wmma_bf16(a, b, acc);
  }
  int mboff = (lane >> 4) * 8;
  float bv = bias[n0 + nl];
  size_t outbase = ((size_t)t * B_SZ + b0 + mboff) * H_SZ + n0 + nl;
#pragma unroll
  for (int r = 0; r < 8; ++r)
    xw[outbase + (size_t)r * H_SZ] = acc[r] + bv;
}

// ---------------------------------------------------------------------------
// proj2: xw[m][n] = sum_k h1[m][k] * Wx2[k][n] + b2[n]   (h1 is bf16 row-major)
// ---------------------------------------------------------------------------
__global__ void __launch_bounds__(256)
proj2_kernel(const __bf16* __restrict__ hin, const __bf16* __restrict__ Wxp,
             const float* __restrict__ bias, float* __restrict__ xw) {
  int tid = threadIdx.x, lane = tid & 31, wave = tid >> 5;
  int tile = blockIdx.x * 8 + wave;
  int mt = tile >> 5;
  int nt = tile & 31;
  int n0 = nt * 16;
  int nl = lane & 15;
  int kb = (lane < 16) ? 0 : 8;
  const __bf16* arow = hin + (size_t)(mt * 16 + nl) * H_SZ;

  v8f acc = {};
#pragma unroll 4
  for (int c = 0; c < H_SZ / 32; ++c) {
    v16bf a = a_frag_from_bf16(arow, c * 32, kb);
    v16bf b = *(const v16bf*)(Wxp + ((size_t)(nt * (H_SZ / 32) + c) * 512 + lane * 16));
    acc = wmma_bf16(a, b, acc);
  }
  int mboff = (lane >> 4) * 8;
  float bv = bias[n0 + nl];
  size_t outbase = (size_t)(mt * 16 + mboff) * H_SZ + n0 + nl;
#pragma unroll
  for (int r = 0; r < 8; ++r)
    xw[outbase + (size_t)r * H_SZ] = acc[r] + bv;
}

// ---------------------------------------------------------------------------
// Persistent recurrent kernel (one block, 32 waves on one WGP):
//   h_t = tanh(xw_t + h_{t-1} @ Wh)
// h_{t-1} lives in LDS (bf16, row-major [32][512]) -> A-fragments are two
// ds_load_b128 each.  Wave w owns output columns [16w, 16w+16): two 16x16
// tiles, 32 WMMAs per step.
//
// Async pipeline: while step t's WMMAs run, the 64 KB xw slab for step t+1 is
// DMA'd into LDS with GLOBAL_LOAD_ASYNC_TO_LDS_B128 (ASYNCcnt); step t's acc
// values are read out of the slab (and a barrier passed) before the re-fill is
// issued, so a single buffer suffices.  First WCACHE_STRIPS column strips of
// packed Wh are also async-staged into LDS to cut per-step L2 traffic.
// ---------------------------------------------------------------------------
#define WCACHE_STRIPS 3
#define HBUF_BYTES   (B_SZ * H_SZ * 2)                 // 32 KB bf16 h state
#define XWBUF_BYTES  (B_SZ * H_SZ * 4)                 // 64 KB f32 xw slab
#define WCACHE_BYTES (WCACHE_STRIPS * 16 * 512 * 2)    // 48 KB Wh strip cache
#define SMEM_BYTES   (HBUF_BYTES + XWBUF_BYTES + WCACHE_BYTES)   // 147456

__global__ void __launch_bounds__(1024)
rnn_kernel(const float* __restrict__ xw, const __bf16* __restrict__ Whp,
           __bf16* __restrict__ hseq, float* __restrict__ hfinal, int T) {
  extern __shared__ char smem[];
  __bf16* hbuf   = (__bf16*)smem;
  float*  xwbuf  = (float*)(smem + HBUF_BYTES);
  __bf16* wcache = (__bf16*)(smem + HBUF_BYTES + XWBUF_BYTES);

  int tid = threadIdx.x, lane = tid & 31, wave = tid >> 5;

  for (int i = tid; i < B_SZ * H_SZ; i += 1024) hbuf[i] = (__bf16)0.0f;

  // Async-stage Wh strip cache and the t=0 xw slab into LDS.
#pragma unroll
  for (int i = 0; i < WCACHE_BYTES / (1024 * 16); ++i) {
    int o = (i * 1024 + tid) * 16;
    async_load_b128((char*)wcache + o, (const char*)Whp + o);
  }
#pragma unroll
  for (int i = 0; i < XWBUF_BYTES / (1024 * 16); ++i) {
    int o = (i * 1024 + tid) * 16;
    async_load_b128((char*)xwbuf + o, (const char*)xw + o);
  }
  wait_asynccnt0();
  __syncthreads();

  int n0 = wave * 16;
  int nl = lane & 15;
  int kb = (lane < 16) ? 0 : 8;
  int mboff = (lane >> 4) * 8;
  const __bf16* wstrip_g = Whp + (size_t)wave * (H_SZ / 32) * 512;
  const __bf16* wstrip_l = wcache + (size_t)wave * (H_SZ / 32) * 512;
  bool useLds = (wave < WCACHE_STRIPS);

  for (int t = 0; t < T; ++t) {
    size_t base = (size_t)t * B_SZ * H_SZ;

    // Pull this step's accumulator init out of the LDS slab.
    v8f acc0, acc1;
#pragma unroll
    for (int r = 0; r < 8; ++r) {
      acc0[r] = xwbuf[(mboff + r) * H_SZ + n0 + nl];
      acc1[r] = xwbuf[(16 + mboff + r) * H_SZ + n0 + nl];
    }
    __syncthreads();  // slab fully consumed -> safe to refill

    // Kick off the async refill for step t+1; overlaps with the WMMAs below.
    if (t + 1 < T) {
      const char* src = (const char*)xw + (base + (size_t)B_SZ * H_SZ) * 4;
#pragma unroll
      for (int i = 0; i < XWBUF_BYTES / (1024 * 16); ++i) {
        int o = (i * 1024 + tid) * 16;
        async_load_b128((char*)xwbuf + o, src + o);
      }
    }

#pragma unroll 4
    for (int c = 0; c < H_SZ / 32; ++c) {
      v16bf b;
      if (useLds) b = *(const v16bf*)(wstrip_l + (size_t)c * 512 + lane * 16);
      else        b = *(const v16bf*)(wstrip_g + (size_t)c * 512 + lane * 16);
      v16bf a0 = a_frag_from_bf16(hbuf + (size_t)nl * H_SZ,        c * 32, kb);
      v16bf a1 = a_frag_from_bf16(hbuf + (size_t)(16 + nl) * H_SZ, c * 32, kb);
      acc0 = wmma_bf16(a0, b, acc0);
      acc1 = wmma_bf16(a1, b, acc1);
    }
    __syncthreads();  // all LDS reads of h_{t-1} complete

#pragma unroll
    for (int r = 0; r < 8; ++r) {
      float v0 = tanhf(acc0[r]);
      float v1 = tanhf(acc1[r]);
      int m0 = mboff + r, m1 = 16 + mboff + r;
      __bf16 q0 = (__bf16)v0, q1 = (__bf16)v1;
      hbuf[(size_t)m0 * H_SZ + n0 + nl] = q0;
      hbuf[(size_t)m1 * H_SZ + n0 + nl] = q1;
      if (hseq) {
        hseq[base + (size_t)m0 * H_SZ + n0 + nl] = q0;
        hseq[base + (size_t)m1 * H_SZ + n0 + nl] = q1;
      }
      if (hfinal && t == T - 1) {
        hfinal[(size_t)m0 * H_SZ + n0 + nl] = v0;
        hfinal[(size_t)m1 * H_SZ + n0 + nl] = v1;
      }
    }
    wait_asynccnt0();  // step t+1 slab has landed in LDS
    __syncthreads();   // h_t + slab visible to all waves
  }
}

// ---------------------------------------------------------------------------
// head: logits = h2 @ Wo + bo ; softmax.  Tiny (32x512x10) -> scalar.
// ---------------------------------------------------------------------------
__global__ void head_kernel(const float* __restrict__ h2, const float* __restrict__ Wo,
                            const float* __restrict__ bo, float* __restrict__ out) {
  int b = threadIdx.x;
  if (b >= B_SZ) return;
  float logits[C_SZ];
#pragma unroll
  for (int c = 0; c < C_SZ; ++c) logits[c] = bo[c];
  for (int k = 0; k < H_SZ; ++k) {
    float hv = h2[b * H_SZ + k];
#pragma unroll
    for (int c = 0; c < C_SZ; ++c) logits[c] = fmaf(hv, Wo[k * C_SZ + c], logits[c]);
  }
  float mx = logits[0];
#pragma unroll
  for (int c = 1; c < C_SZ; ++c) mx = fmaxf(mx, logits[c]);
  float s = 0.0f;
#pragma unroll
  for (int c = 0; c < C_SZ; ++c) { logits[c] = expf(logits[c] - mx); s += logits[c]; }
  float inv = 1.0f / s;
#pragma unroll
  for (int c = 0; c < C_SZ; ++c) out[b * C_SZ + c] = logits[c] * inv;
}

// ---------------------------------------------------------------------------
extern "C" void kernel_launch(void* const* d_in, const int* in_sizes, int n_in,
                              void* d_out, int out_size, void* d_ws, size_t ws_size,
                              hipStream_t stream) {
  (void)in_sizes; (void)n_in; (void)out_size; (void)ws_size;
  const float* x   = (const float*)d_in[0];
  const float* Wx1 = (const float*)d_in[1];
  const float* Wh1 = (const float*)d_in[2];
  const float* b1  = (const float*)d_in[3];
  const float* Wx2 = (const float*)d_in[4];
  const float* Wh2 = (const float*)d_in[5];
  const float* b2  = (const float*)d_in[6];
  const float* Wo  = (const float*)d_in[7];
  const float* bo  = (const float*)d_in[8];
  float* out = (float*)d_out;

  // Workspace layout (xw region reused by both layers)
  char* ws = (char*)d_ws;
  size_t off = 0;
  float*  xw   = (float*)(ws + off);  off += (size_t)M_ROWS * H_SZ * 4;   // 64 MiB
  __bf16* h1   = (__bf16*)(ws + off); off += (size_t)M_ROWS * H_SZ * 2;   // 32 MiB
  __bf16* Wx1p = (__bf16*)(ws + off); off += (size_t)D_SZ * H_SZ * 2;
  __bf16* Wh1p = (__bf16*)(ws + off); off += (size_t)H_SZ * H_SZ * 2;
  __bf16* Wx2p = (__bf16*)(ws + off); off += (size_t)H_SZ * H_SZ * 2;
  __bf16* Wh2p = (__bf16*)(ws + off); off += (size_t)H_SZ * H_SZ * 2;
  float*  h2   = (float*)(ws + off);  off += (size_t)B_SZ * H_SZ * 4;

  // 1) pack weights to bf16 fragment order
  pack_weight_bf16<<<(D_SZ * H_SZ + 255) / 256, 256, 0, stream>>>(Wx1, Wx1p, D_SZ, H_SZ);
  pack_weight_bf16<<<(H_SZ * H_SZ + 255) / 256, 256, 0, stream>>>(Wh1, Wh1p, H_SZ, H_SZ);
  pack_weight_bf16<<<(H_SZ * H_SZ + 255) / 256, 256, 0, stream>>>(Wx2, Wx2p, H_SZ, H_SZ);
  pack_weight_bf16<<<(H_SZ * H_SZ + 255) / 256, 256, 0, stream>>>(Wh2, Wh2p, H_SZ, H_SZ);

  const int tiles = (M_ROWS / 16) * (H_SZ / 16);   // 65536
  // 2) layer-1 input projection (parallel WMMA GEMM)
  proj1_kernel<<<tiles / 8, 256, 0, stream>>>(x, Wx1p, b1, xw);
  // 3) layer-1 recurrence (persistent, async-pipelined, sequential over T)
  rnn_kernel<<<1, 1024, SMEM_BYTES, stream>>>(xw, Wh1p, h1, nullptr, T_SZ);
  // 4) layer-2 input projection (parallel WMMA GEMM), reuse xw
  proj2_kernel<<<tiles / 8, 256, 0, stream>>>(h1, Wx2p, b2, xw);
  // 5) layer-2 recurrence, keep only final h2 (f32)
  rnn_kernel<<<1, 1024, SMEM_BYTES, stream>>>(xw, Wh2p, nullptr, h2, T_SZ);
  // 6) classifier head + softmax
  head_kernel<<<1, 32, 0, stream>>>(h2, Wo, bo, out);
}
